// EmbeddingGNN_12326556139997
// MI455X (gfx1250) — matrix operand order repaired and checked by reading, not compiled, not run on hardware
//
#include <hip/hip_runtime.h>

// ---------------------------------------------------------------------------
// GCN + MLP pipeline for MI455X (gfx1250, wave32, WMMA).
// GEMMs: v_wmma_f32_16x16x32_bf16, branch-free inner loop:
//   A: unconditional float2 vector loads (8B-aligned for all K in this model)
//   B: weights pre-packed into the exact WMMA lane layout -> 1 contiguous
//      32-byte load per lane per tile.
// Segment sums: global_atomic_add_f32.
// ---------------------------------------------------------------------------

typedef __attribute__((ext_vector_type(16))) __bf16 v16bf;
typedef __attribute__((ext_vector_type(8)))  float  v8f;

#define GEMM_BLOCK 128   // 4 waves

// ---------------------------------------------------------------------------
// Weight packing: Bp[((kc*ntilesPad + nt)*32 + lane)*16 + i] = bf16(W[k][n])
//   k = kc*32 + 8*(lane>=16) + (i&7) + 16*(i>>3) ,  n = nt*16 + (lane&15),  zero-padded OOB.
// Matches the ISA 16-bit B-matrix lane layout for V_WMMA_*_16X16X32_BF16.
// ---------------------------------------------------------------------------
__global__ void pack_w_kernel(const float* __restrict__ W, __bf16* __restrict__ Bp,
                              int K, int N, int ntilesPad, int kchunks)
{
    const int idx   = blockIdx.x * blockDim.x + threadIdx.x;
    const int total = kchunks * ntilesPad * 512;
    if (idx >= total) return;
    const int i    = idx & 15;
    const int lane = (idx >> 4) & 31;
    const int r    = idx >> 9;             // kc*ntilesPad + nt
    const int nt   = r % ntilesPad;
    const int kc   = r / ntilesPad;
    const int n    = nt * 16 + (lane & 15);
    const int k    = kc * 32 + ((lane >> 4) << 3) + (i & 7) + ((i >> 3) << 4);
    const float v  = (k < K && n < N) ? W[(size_t)k * N + n] : 0.0f;
    Bp[idx] = (__bf16)v;
}

// ---------------------------------------------------------------------------
// C[M x N] = op(A[M x K]) @ W[K x N] (+bias, optional output relu)
// One wave computes a 16 x 64 tile (4 n-tiles).
// ---------------------------------------------------------------------------
template <bool RELU_IN>
__global__ __launch_bounds__(GEMM_BLOCK)
void gemm_wmma_bf16(const float* __restrict__ A, const __bf16* __restrict__ Bp,
                    const float* __restrict__ bias, float* __restrict__ C,
                    int M, int K, int N, int ntilesPad, int reluOut)
{
    const int wave = threadIdx.x >> 5;
    const int lane = threadIdx.x & 31;
    const int mtile = blockIdx.x;
    const int nt0   = (blockIdx.y * 4 + wave) * 4;   // first of 4 n-tiles
    if (nt0 >= ntilesPad) return;                     // wave-uniform

    const int row   = min(mtile * 16 + (lane & 15), M - 1);
    const int kbase = (lane >> 4) << 3;               // A half-wave K base: 0 or 8
    const int nlane = lane & 15;

    v8f acc[4] = {};

    const float*  __restrict__ Arow = A + (size_t)row * K;
    const float*  __restrict__ pA   = Arow + kbase;
    const __bf16* __restrict__ bp   = Bp + ((size_t)nt0 * 32 + lane) * 16;
    const size_t  strideB           = (size_t)ntilesPad * 512;

    const int kFull = K >> 5;
    const int kcAll = (K + 31) >> 5;

    // ---- full 32-wide K chunks: branch-free vector loads ----
    for (int kc = 0; kc < kFull; ++kc) {
        v16bf a;
#pragma unroll
        for (int h = 0; h < 2; ++h) {                  // runs at kbase+0..7, +16..23
            const float* p = pA + h * 16;
            float2 q0 = *(const float2*)(p + 0);
            float2 q1 = *(const float2*)(p + 2);
            float2 q2 = *(const float2*)(p + 4);
            float2 q3 = *(const float2*)(p + 6);
            if (RELU_IN) {
                q0.x = fmaxf(q0.x, 0.f); q0.y = fmaxf(q0.y, 0.f);
                q1.x = fmaxf(q1.x, 0.f); q1.y = fmaxf(q1.y, 0.f);
                q2.x = fmaxf(q2.x, 0.f); q2.y = fmaxf(q2.y, 0.f);
                q3.x = fmaxf(q3.x, 0.f); q3.y = fmaxf(q3.y, 0.f);
            }
            a[8*h+0] = (__bf16)q0.x; a[8*h+1] = (__bf16)q0.y;
            a[8*h+2] = (__bf16)q1.x; a[8*h+3] = (__bf16)q1.y;
            a[8*h+4] = (__bf16)q2.x; a[8*h+5] = (__bf16)q2.y;
            a[8*h+6] = (__bf16)q3.x; a[8*h+7] = (__bf16)q3.y;
        }
#pragma unroll
        for (int t = 0; t < 4; ++t) {
            const v16bf b = *(const v16bf*)(bp + (size_t)t * 512);
            acc[t] = __builtin_amdgcn_wmma_f32_16x16x32_bf16(
                false, a, false, b, (short)0, acc[t], false, false);
        }
        pA += 32;
        bp += strideB;
    }

    // ---- remainder chunk: clamped-address loads + VALU select (no branches) ----
    if (kFull != kcAll) {
        const int k0 = kFull * 32;
        v16bf a;
#pragma unroll
        for (int j = 0; j < 8; ++j) {
            const int ka = k0 + kbase + j;
            const int kb = ka + 16;
            float va = Arow[min(ka, K - 1)];
            float vb = Arow[min(kb, K - 1)];
            va = (ka < K) ? va : 0.0f;
            vb = (kb < K) ? vb : 0.0f;
            if (RELU_IN) { va = fmaxf(va, 0.f); vb = fmaxf(vb, 0.f); }
            a[j]     = (__bf16)va;
            a[8 + j] = (__bf16)vb;
        }
#pragma unroll
        for (int t = 0; t < 4; ++t) {
            const v16bf b = *(const v16bf*)(bp + (size_t)t * 512);
            acc[t] = __builtin_amdgcn_wmma_f32_16x16x32_bf16(
                false, a, false, b, (short)0, acc[t], false, false);
        }
    }

    // ---- epilogue: C/D layout (VGPR r -> row r + 8*(lane>=16)) ----
    const int rbase = (lane >> 4) << 3;
    for (int t = 0; t < 4; ++t) {
        const int n = (nt0 + t) * 16 + nlane;
        if (n >= N) continue;
        const float bv = bias ? bias[n] : 0.0f;
#pragma unroll
        for (int r = 0; r < 8; ++r) {
            const int rr = mtile * 16 + rbase + r;
            if (rr < M) {
                float v = acc[t][r] + bv;
                if (reluOut) v = fmaxf(v, 0.0f);
                C[(size_t)rr * N + n] = v;
            }
        }
    }
}

// ---------------------------------------------------------------------------
// Graph kernels (F templated -> constant div/mod)
// ---------------------------------------------------------------------------

__global__ void count_deg_kernel(const int* __restrict__ dst, int* __restrict__ cnt, int E)
{
    int e = blockIdx.x * blockDim.x + threadIdx.x;
    if (e < E) atomicAdd(&cnt[dst[e]], 1);
}

__global__ void make_dinv_kernel(const int* __restrict__ cnt, float* __restrict__ dinv, int n)
{
    int i = blockIdx.x * blockDim.x + threadIdx.x;
    if (i < n) dinv[i] = rsqrtf((float)cnt[i] + 1.0f);   // +1 self loop
}

// agg[n][f] = H[n][f] * dinv[n]^2 + bias[f]
template <int F>
__global__ void init_self_kernel(const float* __restrict__ H, const float* __restrict__ dinv,
                                 const float* __restrict__ bias, float* __restrict__ agg, int n)
{
    long long idx = (long long)blockIdx.x * blockDim.x + threadIdx.x;
    if (idx >= (long long)n * F) return;
    const int node = (int)(idx / F);
    const int f    = (int)(idx % F);
    const float di = dinv[node];
    agg[idx] = H[idx] * di * di + bias[f];
}

// agg[dst] += H[src] * dinv[src]*dinv[dst]
template <int F>
__global__ void edge_scatter_kernel(const float* __restrict__ H, const int* __restrict__ src,
                                    const int* __restrict__ dst, const float* __restrict__ dinv,
                                    float* __restrict__ agg, int E)
{
    constexpr int PAIRS = F / 2;
    long long tid = (long long)blockIdx.x * blockDim.x + threadIdx.x;
    if (tid >= (long long)E * PAIRS) return;
    const int e = (int)(tid / PAIRS);
    const int f = (int)(tid % PAIRS) * 2;
    const int s = src[e];
    const int d = dst[e];
    const float w = dinv[s] * dinv[d];
    const float2 h = *(const float2*)(H + (size_t)s * F + f);
    atomicAdd(&agg[(size_t)d * F + f],     h.x * w);
    atomicAdd(&agg[(size_t)d * F + f + 1], h.y * w);
}

__global__ void pool_cnt_kernel(const int* __restrict__ batch, int* __restrict__ cnt, int n)
{
    int i = blockIdx.x * blockDim.x + threadIdx.x;
    if (i < n) atomicAdd(&cnt[batch[i]], 1);
}

// pool[batch[n]][f] += relu(h[n][f])
template <int F>
__global__ void pool_sum_kernel(const float* __restrict__ h, const int* __restrict__ batch,
                                float* __restrict__ pool, int n)
{
    long long idx = (long long)blockIdx.x * blockDim.x + threadIdx.x;
    if (idx >= (long long)n * F) return;
    const int node = (int)(idx / F);
    const int f    = (int)(idx % F);
    atomicAdd(&pool[(size_t)batch[node] * F + f], fmaxf(h[idx], 0.0f));
}

template <int F>
__global__ void pool_div_kernel(float* __restrict__ pool, const int* __restrict__ cnt, int G)
{
    long long idx = (long long)blockIdx.x * blockDim.x + threadIdx.x;
    if (idx >= (long long)G * F) return;
    const int g = (int)(idx / F);
    pool[idx] = pool[idx] / (float)max(cnt[g], 1);
}

// ---------------------------------------------------------------------------
// Host-side launch
// ---------------------------------------------------------------------------

static inline size_t alignUp256(size_t x) { return (x + 255) & ~(size_t)255; }
static inline int ntilesPadOf(int N) { return ((((N + 15) / 16) + 3) / 4) * 4; }
static inline int kchunksOf(int K)   { return (K + 31) / 32; }

static inline void launch_pack(const float* W, __bf16* Bp, int K, int N, hipStream_t stream)
{
    const int ntp = ntilesPadOf(N), kc = kchunksOf(K);
    const int total = kc * ntp * 512;
    pack_w_kernel<<<(total + 255) / 256, 256, 0, stream>>>(W, Bp, K, N, ntp, kc);
}

template <bool RELU_IN>
static inline void launch_gemm(const float* A, const __bf16* Bp, const float* bias, float* C,
                               int M, int K, int N, int reluOut, hipStream_t stream)
{
    const int ntp    = ntilesPadOf(N);
    const int mtiles = (M + 15) / 16;
    const int gy     = (ntp / 4 + 3) / 4;    // 4 waves/block, 4 n-tiles/wave
    dim3 grid(mtiles, gy, 1);
    gemm_wmma_bf16<RELU_IN><<<grid, GEMM_BLOCK, 0, stream>>>(A, Bp, bias, C, M, K, N, ntp, reluOut);
}

extern "C" void kernel_launch(void* const* d_in, const int* in_sizes, int n_in,
                              void* d_out, int out_size, void* d_ws, size_t ws_size,
                              hipStream_t stream)
{
    (void)in_sizes; (void)n_in; (void)out_size; (void)ws_size;

    constexpr int NN = 500000;   // nodes
    constexpr int NE = 1000000;  // edges
    constexpr int NG = 20000;    // graphs
    constexpr int F1 = 78, F2 = 156, F3 = 312, FH = 1024, FO = 128;

    const float* x     = (const float*)d_in[0];
    const int*   ei    = (const int*)d_in[1];
    const int*   batch = (const int*)d_in[2];
    const float* W1    = (const float*)d_in[3];
    const float* b1    = (const float*)d_in[4];
    const float* W2    = (const float*)d_in[5];
    const float* b2    = (const float*)d_in[6];
    const float* W3    = (const float*)d_in[7];
    const float* b3    = (const float*)d_in[8];
    const float* fW1   = (const float*)d_in[9];
    const float* fb1   = (const float*)d_in[10];
    const float* fW2   = (const float*)d_in[11];
    const float* fb2   = (const float*)d_in[12];
    float*       out   = (float*)d_out;

    const int* src = ei;            // edge_index[0]
    const int* dst = ei + NE;       // edge_index[1]

    // ---- workspace carve-up ----
    char* w = (char*)d_ws;
    int*   cntNode = (int*)w;   w += alignUp256((size_t)NN * sizeof(int));
    float* dinv    = (float*)w; w += alignUp256((size_t)NN * sizeof(float));
    int*   gcnt    = (int*)w;   w += alignUp256((size_t)NG * sizeof(int));
    float* R0      = (float*)w; w += alignUp256((size_t)NN * F3 * sizeof(float));
    float* R1      = (float*)w; w += alignUp256((size_t)NN * F3 * sizeof(float));
    float* R2      = (float*)w; w += alignUp256((size_t)NN * F2 * sizeof(float));
    float* pool    = (float*)w; w += alignUp256((size_t)NG * F3 * sizeof(float));
    float* f1buf   = (float*)w; w += alignUp256((size_t)NG * FH * sizeof(float));
    __bf16* Bp1  = (__bf16*)w; w += alignUp256((size_t)kchunksOf(F1) * ntilesPadOf(F1) * 512 * 2);
    __bf16* Bp2  = (__bf16*)w; w += alignUp256((size_t)kchunksOf(F1) * ntilesPadOf(F2) * 512 * 2);
    __bf16* Bp3  = (__bf16*)w; w += alignUp256((size_t)kchunksOf(F2) * ntilesPadOf(F3) * 512 * 2);
    __bf16* BpF1 = (__bf16*)w; w += alignUp256((size_t)kchunksOf(F3) * ntilesPadOf(FH) * 512 * 2);
    __bf16* BpF2 = (__bf16*)w; w += alignUp256((size_t)kchunksOf(FH) * ntilesPadOf(FO) * 512 * 2);

    const int TB = 256;
    auto blocksFor = [](long long n, int tb) { return (unsigned)((n + tb - 1) / tb); };

    // ---- pack all weights into WMMA B lane layout (tiny) ----
    launch_pack(W1,  Bp1,  F1, F1, stream);
    launch_pack(W2,  Bp2,  F1, F2, stream);
    launch_pack(W3,  Bp3,  F2, F3, stream);
    launch_pack(fW1, BpF1, F3, FH, stream);
    launch_pack(fW2, BpF2, FH, FO, stream);

    // ---- degrees: deg = 1 + count(dst); dinv = rsqrt(deg) ----
    hipMemsetAsync(cntNode, 0, (size_t)NN * sizeof(int), stream);
    count_deg_kernel<<<blocksFor(NE, TB), TB, 0, stream>>>(dst, cntNode, NE);
    make_dinv_kernel<<<blocksFor(NN, TB), TB, 0, stream>>>(cntNode, dinv, NN);

    // ---- GCN layer 1: H = x @ W1; agg1 = scatter + self + b1 (relu deferred) ----
    launch_gemm<false>(x, Bp1, nullptr, R0, NN, F1, F1, 0, stream);
    init_self_kernel<F1><<<blocksFor((long long)NN * F1, TB), TB, 0, stream>>>(R0, dinv, b1, R1, NN);
    edge_scatter_kernel<F1><<<blocksFor((long long)NE * (F1 / 2), TB), TB, 0, stream>>>(
        R0, src, dst, dinv, R1, NE);

    // ---- GCN layer 2: H = relu(agg1) @ W2 ----
    launch_gemm<true>(R1, Bp2, nullptr, R0, NN, F1, F2, 0, stream);
    init_self_kernel<F2><<<blocksFor((long long)NN * F2, TB), TB, 0, stream>>>(R0, dinv, b2, R2, NN);
    edge_scatter_kernel<F2><<<blocksFor((long long)NE * (F2 / 2), TB), TB, 0, stream>>>(
        R0, src, dst, dinv, R2, NE);

    // ---- GCN layer 3: H = relu(agg2) @ W3 ----
    launch_gemm<true>(R2, Bp3, nullptr, R0, NN, F2, F3, 0, stream);
    init_self_kernel<F3><<<blocksFor((long long)NN * F3, TB), TB, 0, stream>>>(R0, dinv, b3, R1, NN);
    edge_scatter_kernel<F3><<<blocksFor((long long)NE * (F3 / 2), TB), TB, 0, stream>>>(
        R0, src, dst, dinv, R1, NE);

    // ---- global mean pool over relu(agg3) ----
    hipMemsetAsync(pool, 0, (size_t)NG * F3 * sizeof(float), stream);
    hipMemsetAsync(gcnt, 0, (size_t)NG * sizeof(int), stream);
    pool_cnt_kernel<<<blocksFor(NN, TB), TB, 0, stream>>>(batch, gcnt, NN);
    pool_sum_kernel<F3><<<blocksFor((long long)NN * F3, TB), TB, 0, stream>>>(R1, batch, pool, NN);
    pool_div_kernel<F3><<<blocksFor((long long)NG * F3, TB), TB, 0, stream>>>(pool, gcnt, NG);

    // ---- MLP head ----
    launch_gemm<false>(pool, BpF1, fb1, f1buf, NG, F3, FH, /*reluOut=*/1, stream);
    launch_gemm<false>(f1buf, BpF2, fb2, out, NG, FH, FO, /*reluOut=*/0, stream);
}
